// CrossAttentionFromSelf_23295902613627
// MI455X (gfx1250) — matrix operand order repaired and checked by reading, not compiled, not run on hardware
//
#include <hip/hip_runtime.h>
#include <hip/hip_bf16.h>

// ---------------------------------------------------------------------------
// CrossAttentionFromSelf for MI455X (gfx1250, wave32, WMMA f16 16x16x32)
// B=2, S=2048, D=2048, H=16, HD=128
// Round 3: vectorized f32->f16 staging (keep global_load_b128 alive via
//          __builtin_convertvector) + packed b32 stores for the V transpose.
// ---------------------------------------------------------------------------

typedef __attribute__((ext_vector_type(16))) _Float16     v16h;
typedef __attribute__((ext_vector_type(4)))  _Float16     h4;
typedef __attribute__((ext_vector_type(8)))  float        v8f;
typedef __attribute__((ext_vector_type(4)))  unsigned int u4;
typedef __attribute__((ext_vector_type(8)))  int          i8v;
typedef __attribute__((ext_vector_type(4)))  int          i4v;
typedef __attribute__((ext_vector_type(4)))  float        f4;

#define S_LEN 2048
#define D_DIM 2048
#define NHEAD 16
#define HEADD 128

__device__ __forceinline__ v8f wmma16(v16h a, v16h b, v8f c) {
  return __builtin_amdgcn_wmma_f32_16x16x32_f16(
      /*neg_a=*/false, a, /*neg_b=*/false, b,
      /*c_mod=*/(short)0, c, /*reuse_a=*/false, /*reuse_b=*/false);
}

// A-fragment (16x32 f16): lanes 0-15 hold M=lane, K={0..7,16..23};
// lanes 16-31 hold M=lane-16, K={8..15,24..31}.
__device__ __forceinline__ v16h lds_frag_a(const _Float16* p0, int lane, int ld) {
  const int mm = lane & 15, hf = lane >> 4;
  const _Float16* p = p0 + mm * ld + hf * 8;
  v16h r;
  ((u4*)&r)[0] = *(const u4*)(p);
  ((u4*)&r)[1] = *(const u4*)(p + 16);
  return r;
}

// B-fragment (32x16 f16): lane n holds column n; lanes 0-15 K={0..7,16..23},
// lanes 16-31 K={8..15,24..31} (B stored transposed).
__device__ __forceinline__ v16h lds_frag_b(const _Float16* p0, int lane, int ld) {
  const int nn = lane & 15, hf = lane >> 4;
  const _Float16* p = p0 + nn * ld + hf * 8;
  v16h r;
  ((u4*)&r)[0] = *(const u4*)(p);
  ((u4*)&r)[1] = *(const u4*)(p + 16);
  return r;
}

// 16 f32 -> 16 f16, keeping the b128 loads vector-shaped so SROA does not
// scalarize them into global_load_b32 x4.
__device__ __forceinline__ void cvt_f32_to_f16_16(const float* src, _Float16* dst) {
#pragma unroll
  for (int j = 0; j < 4; ++j) {
    f4 v = *(const f4*)(src + j * 4);
    *(h4*)(dst + j * 4) = __builtin_convertvector(v, h4);
  }
}

// low 32 bits of a generic LDS pointer = LDS byte address (aperture form)
__device__ __forceinline__ unsigned lds_off(const void* p) {
  return (unsigned)(unsigned long long)(size_t)p;
}

// ---------------------------------------------------------------------------
// Tensor Data Mover: 2D f16 tile load, global(row-major, stride0) -> LDS with
// row padding. tile_d0 = contiguous f16 per row, tile_d1 = rows.
// pad_icode: DWORDs-before-pad code (3=16DW, 5=64DW); pad_acode: pad DWORDs-1.
// D# layout per CDNA5 ISA (2D: groups 2,3 = 0).
// ---------------------------------------------------------------------------
__device__ __forceinline__ void tdm_load_2d_f16(
    unsigned lds_byte_addr, const void* gsrc, unsigned tile_d0, unsigned tile_d1,
    unsigned stride0, unsigned pad_icode, unsigned pad_acode) {
  const unsigned long long ga = (unsigned long long)(size_t)gsrc;
  u4 g0;
  g0[0] = 1u;                                                // count=1
  g0[1] = lds_byte_addr;                                     // lds_addr
  g0[2] = (unsigned)(ga & 0xffffffffu);                      // global_addr[31:0]
  g0[3] = (unsigned)((ga >> 32) & 0x1ffffffu) | (2u << 30);  // ga[56:32] | type=2
  i8v g1;
  g1[0] = (int)((1u << 16) |                                 // data_size = 2 bytes
                (1u << 20) |                                 // pad_enable
                (pad_icode << 22) | (pad_acode << 25));
  g1[1] = (int)((tile_d0 & 0xffffu) << 16);                  // tensor_dim0[15:0]
  g1[2] = (int)(((tile_d0 >> 16) & 0xffffu) |                // tensor_dim0[31:16]
                ((tile_d1 & 0xffffu) << 16));                // tensor_dim1[15:0]
  g1[3] = (int)(((tile_d1 >> 16) & 0xffffu) |                // tensor_dim1[31:16]
                ((tile_d0 & 0xffffu) << 16));                // tile_dim0
  g1[4] = (int)(tile_d1 & 0xffffu);                          // tile_dim1, tile_dim2=0
  g1[5] = (int)stride0;                                      // dim0_stride[31:0]
  g1[6] = 0;
  g1[7] = 0;
  const i4v z4 = {0, 0, 0, 0};
#if __clang_major__ >= 23
  const i8v z8 = {0, 0, 0, 0, 0, 0, 0, 0};
  __builtin_amdgcn_tensor_load_to_lds(g0, g1, z4, z4, z8, 0);
#else
  __builtin_amdgcn_tensor_load_to_lds(g0, g1, z4, z4, 0);
#endif
}

// ---------------------------------------------------------------------------
// GEMM: C[M x 2048] = A[M x 2048] * W[2048 x 2048]^T + bias
// 128x128 block tile, 256 threads (8 waves), BK=32, double-buffered LDS.
// AF16: A is f16 in global (staged by TDM). OUTF32: write float (else f16).
// ---------------------------------------------------------------------------
template <bool AF16, bool OUTF32>
__global__ __launch_bounds__(256) void gemm_xwT(
    const void* __restrict__ Ain, const float* __restrict__ W,
    const float* __restrict__ bias, void* __restrict__ Cout) {
  constexpr int K = D_DIM;
  constexpr int N = D_DIM;
  constexpr int LDT = 40;  // padded LDS row stride (f16): 16 DW data + 4 DW pad

  __shared__ _Float16 As[2][128 * LDT];
  __shared__ _Float16 Bs[2][128 * LDT];

  const int t = threadIdx.x;
  const int lane = t & 31;
  const int wvi = t >> 5;
  const int wm = wvi & 3;   // 4 strips of 32 rows
  const int wn = wvi >> 2;  // 2 strips of 64 cols
  const int m0 = blockIdx.y * 128;
  const int n0 = blockIdx.x * 128;
  const int lrow = t >> 1;        // staging row 0..127
  const int lcg = (t & 1) * 16;   // staging col group

  const float* wSrc = W + (size_t)(n0 + lrow) * K + lcg;
  const float* aSrcF = (const float*)Ain + (size_t)(m0 + lrow) * K + lcg;
  const _Float16* aBaseH = (const _Float16*)Ain + (size_t)m0 * K;

  v8f acc[2][4];
#pragma unroll
  for (int i = 0; i < 2; ++i)
#pragma unroll
    for (int j = 0; j < 4; ++j) acc[i][j] = {};

  // ---- prologue: stage k0 = 0 into buffer 0 ----
  {
    alignas(16) _Float16 wreg[16];
    cvt_f32_to_f16_16(wSrc, wreg);
    *(u4*)&Bs[0][lrow * LDT + lcg] = ((u4*)wreg)[0];
    *(u4*)&Bs[0][lrow * LDT + lcg + 8] = ((u4*)wreg)[1];
    if (AF16) {
      if (wvi == 0) {
        tdm_load_2d_f16(lds_off(&As[0][0]), aBaseH, 32, 128, K, 3, 3);
        __builtin_amdgcn_s_wait_tensorcnt(0);
      }
    } else {
      alignas(16) _Float16 areg[16];
      cvt_f32_to_f16_16(aSrcF, areg);
      *(u4*)&As[0][lrow * LDT + lcg] = ((u4*)areg)[0];
      *(u4*)&As[0][lrow * LDT + lcg + 8] = ((u4*)areg)[1];
    }
  }
  __syncthreads();

  int buf = 0;
  for (int k0 = 0; k0 < K; k0 += 32) {
    const int kn = k0 + 32;
    const bool nxt = kn < K;
    // ---- issue next tile's loads while current buffer computes ----
    alignas(16) _Float16 wreg[16];
    alignas(16) _Float16 areg[16];
    if (nxt) {
      cvt_f32_to_f16_16(wSrc + kn, wreg);
      if (AF16) {
        if (wvi == 0)
          tdm_load_2d_f16(lds_off(&As[buf ^ 1][0]), aBaseH + kn, 32, 128, K, 3, 3);
      } else {
        cvt_f32_to_f16_16(aSrcF + kn, areg);
      }
    }
    // ---- compute from current buffer: 2 (M) x 4 (N) WMMA tiles / wave ----
    v16h af[2], bfv[4];
#pragma unroll
    for (int i = 0; i < 2; ++i)
      af[i] = lds_frag_a(&As[buf][(wm * 32 + i * 16) * LDT], lane, LDT);
#pragma unroll
    for (int j = 0; j < 4; ++j)
      bfv[j] = lds_frag_b(&Bs[buf][(wn * 64 + j * 16) * LDT], lane, LDT);
#pragma unroll
    for (int i = 0; i < 2; ++i)
#pragma unroll
      for (int j = 0; j < 4; ++j) acc[i][j] = wmma16(af[i], bfv[j], acc[i][j]);
    // ---- commit next tile into the alternate buffer ----
    if (nxt) {
      *(u4*)&Bs[buf ^ 1][lrow * LDT + lcg] = ((u4*)wreg)[0];
      *(u4*)&Bs[buf ^ 1][lrow * LDT + lcg + 8] = ((u4*)wreg)[1];
      if (AF16) {
        if (wvi == 0) __builtin_amdgcn_s_wait_tensorcnt(0);
      } else {
        *(u4*)&As[buf ^ 1][lrow * LDT + lcg] = ((u4*)areg)[0];
        *(u4*)&As[buf ^ 1][lrow * LDT + lcg + 8] = ((u4*)areg)[1];
      }
    }
    __syncthreads();
    buf ^= 1;
  }

  // ---- epilogue: C layout — VGPR e, lane: row = e + 8*(lane>=16), col = lane&15
  const int hf = lane >> 4, cn = lane & 15;
#pragma unroll
  for (int j = 0; j < 4; ++j) {
    const int col = n0 + wn * 64 + j * 16 + cn;
    const float bvv = bias[col];
#pragma unroll
    for (int i = 0; i < 2; ++i) {
#pragma unroll
      for (int e = 0; e < 8; ++e) {
        const int row = m0 + wm * 32 + i * 16 + e + hf * 8;
        const float v = acc[i][j][e] + bvv;
        if (OUTF32)
          ((float*)Cout)[(size_t)row * N + col] = v;
        else
          ((_Float16*)Cout)[(size_t)row * N + col] = (_Float16)v;
      }
    }
  }
}

// ---------------------------------------------------------------------------
// RoPE in-place on f16 [B,S,D] buffer (e = h*128 + d), pairs (i, i+64).
// ---------------------------------------------------------------------------
__global__ __launch_bounds__(256) void rope_kernel(_Float16* __restrict__ x) {
  const int t = blockIdx.x * blockDim.x + threadIdx.x;
  const int i = t & 63;
  const int h = (t >> 6) & 15;
  const int s = (t >> 10) & 2047;
  const int b = t >> 21;
  const size_t base = ((size_t)(b * S_LEN + s)) * D_DIM + h * HEADD;
  const float inv = __powf(10000.0f, -(float)i * (1.0f / 64.0f));
  const float ang = (float)s * inv;
  float c, sn;
  __sincosf(ang, &c, &sn);
  const float x1 = (float)x[base + i];
  const float x2 = (float)x[base + i + 64];
  x[base + i] = (_Float16)(x1 * c - x2 * sn);
  x[base + i + 64] = (_Float16)(x2 * c + x1 * sn);
}

// ---------------------------------------------------------------------------
// Flash attention: grid (S/64, B*H), 128 threads = 4 waves.
// K block staged by TDM (64x128 f16 tile, padded rows); V staged transposed
// with packed key-pair b32 DS stores.
// ---------------------------------------------------------------------------
__global__ __launch_bounds__(128) void attn_kernel(
    const _Float16* __restrict__ qf, const _Float16* __restrict__ kf,
    const _Float16* __restrict__ vf, const float* __restrict__ mask,
    _Float16* __restrict__ of) {
  constexpr int LDK = 136;  // 64 DW data + 4 DW pad per row
  constexpr int LDV = 72;

  __shared__ _Float16 Ks[64 * LDK];     // K block, row-major [key][d]
  __shared__ _Float16 Vt[128 * LDV];    // V block, transposed [d][key]
  __shared__ _Float16 Ps[4][16 * LDV];  // per-wave P scratch [q][key]

  const int t = threadIdx.x;
  const int lane = t & 31;
  const int wv = t >> 5;
  const int mm = lane & 15, hf = lane >> 4, cn = lane & 15;
  const int q0 = blockIdx.x * 64;
  const int bh = blockIdx.y;
  const int b = bh >> 4, h = bh & 15;

  // --- Q A-fragments for this wave's 16 rows, straight from global ---
  v16h qa[4];
  {
    const _Float16* p =
        qf + (size_t)(b * S_LEN + q0 + wv * 16 + mm) * D_DIM + h * HEADD;
#pragma unroll
    for (int ks = 0; ks < 4; ++ks) {
      ((u4*)&qa[ks])[0] = *(const u4*)(p + ks * 32 + hf * 8);
      ((u4*)&qa[ks])[1] = *(const u4*)(p + ks * 32 + 16 + hf * 8);
    }
  }

  float mrow[8], lrow[8];
  v8f accO[8];
#pragma unroll
  for (int e = 0; e < 8; ++e) { mrow[e] = -3.0e38f; lrow[e] = 0.0f; }
#pragma unroll
  for (int dt = 0; dt < 8; ++dt) accO[dt] = {};

  const float scale = 0.08838834764831845f;  // 1/sqrt(128)

  for (int kb = 0; kb < S_LEN; kb += 64) {
    __syncthreads();
    // --- K block via Tensor Data Mover (one wave issues for the group) ---
    if (wv == 0)
      tdm_load_2d_f16(lds_off(&Ks[0]),
                      kf + (size_t)(b * S_LEN + kb) * D_DIM + h * HEADD,
                      /*d0=*/128, /*d1=*/64, /*stride0=*/D_DIM,
                      /*pad_i=*/5, /*pad_a=*/3);
    // --- V block staged transposed: each thread stages a key pair over a
    //     32-wide d group; adjacent keys pack into one b32 DS store ---
    {
      const int kp = (t & 31) * 2;   // keys kp, kp+1
      const int cg = (t >> 5) * 32;  // d col group of 32
      const _Float16* vp =
          vf + (size_t)(b * S_LEN + kb + kp) * D_DIM + h * HEADD + cg;
#pragma unroll
      for (int j8 = 0; j8 < 4; ++j8) {
        alignas(16) _Float16 v0[8], v1[8];
        *(u4*)v0 = *(const u4*)(vp + j8 * 8);
        *(u4*)v1 = *(const u4*)(vp + D_DIM + j8 * 8);
#pragma unroll
        for (int j = 0; j < 8; ++j) {
          const unsigned lo = *(const unsigned short*)&v0[j];
          const unsigned hi = *(const unsigned short*)&v1[j];
          *(unsigned*)&Vt[(cg + j8 * 8 + j) * LDV + kp] = lo | (hi << 16);
        }
      }
    }
    if (wv == 0) __builtin_amdgcn_s_wait_tensorcnt(0);
    __syncthreads();

    // --- S = Q K^T (scale + mask), 4 key sub-tiles of 16 ---
    float st[4][8];
#pragma unroll
    for (int nt = 0; nt < 4; ++nt) {
      v8f s = {};
#pragma unroll
      for (int ks = 0; ks < 4; ++ks) {
        v16h bfr = lds_frag_b(&Ks[(nt * 16) * LDK + ks * 32], lane, LDK);
        s = wmma16(qa[ks], bfr, s);
      }
#pragma unroll
      for (int e = 0; e < 8; ++e) {
        const int qr = q0 + wv * 16 + e + hf * 8;
        const float mv =
            mask[(size_t)(b * S_LEN + qr) * S_LEN + kb + nt * 16 + cn];
        st[nt][e] = s[e] * scale + mv;
      }
    }

    // --- online softmax (rows live across 16-lane halves) ---
    float rmax[8];
#pragma unroll
    for (int e = 0; e < 8; ++e) {
      float r = st[0][e];
#pragma unroll
      for (int nt = 1; nt < 4; ++nt) r = fmaxf(r, st[nt][e]);
      rmax[e] = r;
    }
#pragma unroll
    for (int off = 1; off < 16; off <<= 1)
#pragma unroll
      for (int e = 0; e < 8; ++e)
        rmax[e] = fmaxf(rmax[e], __shfl_xor(rmax[e], off, 32));

    float alpha[8], rsum[8];
#pragma unroll
    for (int e = 0; e < 8; ++e) {
      const float mn = fmaxf(mrow[e], rmax[e]);
      alpha[e] = __expf(mrow[e] - mn);
      mrow[e] = mn;
      float rs = 0.0f;
#pragma unroll
      for (int nt = 0; nt < 4; ++nt) {
        st[nt][e] = __expf(st[nt][e] - mn);
        rs += st[nt][e];
      }
      rsum[e] = rs;
    }
#pragma unroll
    for (int off = 1; off < 16; off <<= 1)
#pragma unroll
      for (int e = 0; e < 8; ++e) rsum[e] += __shfl_xor(rsum[e], off, 32);
#pragma unroll
    for (int e = 0; e < 8; ++e) lrow[e] = lrow[e] * alpha[e] + rsum[e];
#pragma unroll
    for (int dt = 0; dt < 8; ++dt)
#pragma unroll
      for (int e = 0; e < 8; ++e) accO[dt][e] *= alpha[e];

    // --- spill P (C layout) into per-wave LDS scratch as f16 ---
#pragma unroll
    for (int nt = 0; nt < 4; ++nt)
#pragma unroll
      for (int e = 0; e < 8; ++e)
        Ps[wv][(e + hf * 8) * LDV + nt * 16 + cn] = (_Float16)st[nt][e];
    asm volatile("s_wait_dscnt 0" ::: "memory");  // intra-wave LDS turnaround

    // --- O += P * V : two K=32 steps over 64 keys, 8 d-tiles ---
#pragma unroll
    for (int kq = 0; kq < 2; ++kq) {
      v16h pa = lds_frag_a(&Ps[wv][kq * 32], lane, LDV);
#pragma unroll
      for (int dt = 0; dt < 8; ++dt) {
        v16h bvf = lds_frag_b(&Vt[(dt * 16) * LDV + kq * 32], lane, LDV);
        accO[dt] = wmma16(pa, bvf, accO[dt]);
      }
    }
  }

  // --- normalize and write O (f16, [B,S,D] layout) ---
#pragma unroll
  for (int dt = 0; dt < 8; ++dt) {
#pragma unroll
    for (int e = 0; e < 8; ++e) {
      const int qr = q0 + wv * 16 + e + hf * 8;
      of[(size_t)(b * S_LEN + qr) * D_DIM + h * HEADD + dt * 16 + cn] =
          (_Float16)(accO[dt][e] * (1.0f / lrow[e]));
    }
  }
}

// ---------------------------------------------------------------------------
// Host launcher
// ---------------------------------------------------------------------------
extern "C" void kernel_launch(void* const* d_in, const int* in_sizes, int n_in,
                              void* d_out, int out_size, void* d_ws,
                              size_t ws_size, hipStream_t stream) {
  (void)in_sizes; (void)n_in; (void)out_size; (void)ws_size;
  const float* query = (const float*)d_in[0];
  const float* keyv  = (const float*)d_in[1];
  const float* mask  = (const float*)d_in[2];
  const float* Wq = (const float*)d_in[3];
  const float* bq = (const float*)d_in[4];
  const float* Wk = (const float*)d_in[5];
  const float* bk = (const float*)d_in[6];
  const float* Wv = (const float*)d_in[7];
  const float* bv = (const float*)d_in[8];
  const float* Wo = (const float*)d_in[9];
  const float* bo = (const float*)d_in[10];

  const size_t MAT = (size_t)2 * S_LEN * D_DIM;  // 4096*2048 elements
  _Float16* qf = (_Float16*)d_ws;
  _Float16* kf = qf + MAT;
  _Float16* vf = kf + MAT;
  _Float16* of = vf + MAT;

  const dim3 gemm_grid(D_DIM / 128, (2 * S_LEN) / 128);  // (16, 32)
  gemm_xwT<false, false><<<gemm_grid, 256, 0, stream>>>(query, Wq, bq, qf);
  gemm_xwT<false, false><<<gemm_grid, 256, 0, stream>>>(keyv,  Wk, bk, kf);
  gemm_xwT<false, false><<<gemm_grid, 256, 0, stream>>>(keyv,  Wv, bv, vf);

  const int rope_threads = 2 * S_LEN * NHEAD * 64;  // 2^22
  rope_kernel<<<rope_threads / 256, 256, 0, stream>>>(qf);
  rope_kernel<<<rope_threads / 256, 256, 0, stream>>>(kf);

  attn_kernel<<<dim3(S_LEN / 64, 2 * NHEAD), 128, 0, stream>>>(qf, kf, vf, mask, of);

  gemm_xwT<true, true><<<gemm_grid, 256, 0, stream>>>(of, Wo, bo, (float*)d_out);
}
